// Net_71451075936316
// MI455X (gfx1250) — matrix operand order repaired and checked by reading, not compiled, not run on hardware
//
#include <hip/hip_runtime.h>

typedef __attribute__((ext_vector_type(16))) _Float16 v16h;
typedef __attribute__((ext_vector_type(8)))  _Float16 v8h;
typedef __attribute__((ext_vector_type(8)))  float    v8f;

// ---------------- geometry ----------------
#define NHG     63
#define NPATCH  3969          // 63*63 patches
// LDS: s_in 2*1024*2B=4096 | act1t 625*32*2B=40000 (transposed, ch-padded, reused by stage3) |
//      act2t 841*64*2B=107648 (29x29 zero halo, ch-padded) | cwt 64*24*4B=6144
#define OFF_ACT1T 4096
#define OFF_ACT2T (4096 + 40000)             // 44096
#define OFF_CWT   (4096 + 40000 + 107648)    // 151744
#define SMEM_BYTES (OFF_CWT + 6144)          // 157888

// workspace layout (bytes)
#define WS_RECON 0u                        // 1024*1024 f32 = 4 MB
#define WS_W1H   (4u*1024u*1024u)          // conv1 weights f16, 32 x 128 (k = kd*64+ky*8+kx)
#define WS_W2H   (WS_W1H + 32u*128u*2u)    // conv2 weights f16, 64 x 800 (k = tap*32+ic)
#define WS_W2T   (WS_W2H + 64u*800u*2u)    // deconv2-as-conv f16, 32 x 1600 (k = tap*64+oc)
#define WS_W1T   (WS_W2T + 32u*1600u*2u)   // deconv1 flipped f32, 24 x 2 x 64

__device__ __forceinline__ float elu_f(float x) { return x > 0.f ? x : (__expf(x) - 1.f); }

// A fragment for v_wmma_f32_16x16x32_f16 (this code's contiguous K order, B uses the same):
// lane<16: row M=lane,    K = {0..15} of this K-chunk
// lane>=16: row M=lane-16, K = {16..31}
__device__ __forceinline__ v16h a_frag(const _Float16* __restrict__ A, int ldk,
                                       int mbase, int kbase) {
  const int lane = threadIdx.x & 31;
  const _Float16* p = A + (mbase + (lane & 15)) * ldk + kbase + ((lane >> 4) << 4);
  v16h r;
  ((v8h*)&r)[0] = *(const v8h*)(p);        // two contiguous 16B loads
  ((v8h*)&r)[1] = *(const v8h*)(p + 8);
  return r;
}

// B fragment when the lane's 16 K-values are contiguous in LDS.
__device__ __forceinline__ v16h b_frag_lds(const _Float16* p) {
  v16h r;
  ((v8h*)&r)[0] = *(const v8h*)(p);        // ds_load_b128
  ((v8h*)&r)[1] = *(const v8h*)(p + 8);    // ds_load_b128
  return r;
}

#define WMMA_F16(a, b, c) \
  __builtin_amdgcn_wmma_f32_16x16x32_f16(false, (a), false, (b), (short)0, (c), false, false)

// ---------------- kernel 1: zero overlap-add accumulator ----------------
__global__ void zero_recon(float* __restrict__ r) {
  r[blockIdx.x * 256 + threadIdx.x] = 0.f;   // grid = 4096 blocks exactly
}

// ---------------- kernel 2: pack weights (K-permuted to match LDS layouts) ----
__global__ void prep_weights(const float* __restrict__ c1w, const float* __restrict__ c2w,
                             const float* __restrict__ d2w, const float* __restrict__ d1w,
                             _Float16* __restrict__ w1h, _Float16* __restrict__ w2h,
                             _Float16* __restrict__ w2t, float* __restrict__ w1t) {
  const int tid = blockIdx.x * blockDim.x + threadIdx.x;
  const int str = gridDim.x * blockDim.x;
  // conv1: (24,1,2,8,8) -> [32][128], k = kd*64+ky*8+kx (flat copy), pad rows 0
  for (int i = tid; i < 32 * 128; i += str) {
    int m = i >> 7;
    w1h[i] = (m < 24) ? (_Float16)c1w[i] : (_Float16)0.f;
  }
  // conv2: (60,24,5,5) -> [64][800], k = (ky*5+kx)*32 + ic, pad 0
  for (int i = tid; i < 64 * 800; i += str) {
    int m = i / 800, k = i % 800;
    int tap = k >> 5, ic = k & 31, ky = tap / 5, kx = tap % 5;
    w2h[i] = (m < 60 && ic < 24) ? (_Float16)c2w[m * 600 + ic * 25 + ky * 5 + kx]
                                 : (_Float16)0.f;
  }
  // deconv2 as conv: [32][1600], k = (ky*5+kx)*64 + oc; w = d2w[oc][m][4-ky][4-kx]
  for (int i = tid; i < 32 * 1600; i += str) {
    int m = i / 1600, k = i % 1600;
    int tap = k >> 6, oc = k & 63, ky = tap / 5, kx = tap % 5;
    float v = 0.f;
    if (m < 24 && oc < 60)
      v = d2w[oc * 600 + m * 25 + (4 - ky) * 5 + (4 - kx)];
    w2t[i] = (_Float16)v;
  }
  // deconv1 flipped: w1t[ic][od][ky*8+kx] = d1w[ic][0][od][7-ky][7-kx]
  for (int i = tid; i < 24 * 2 * 64; i += str) {
    int ic = i >> 7, r = i & 127, od = r >> 6, kk = r & 63;
    int ky = kk >> 3, kx = kk & 7;
    w1t[i] = d1w[ic * 128 + od * 64 + (7 - ky) * 8 + (7 - kx)];
  }
}

// ---------------- kernel 3: per-patch LDS-resident autoencoder pipeline ----
__global__ __launch_bounds__(256) void patch_pipeline(
    const float* __restrict__ x1, const float* __restrict__ x2,
    const _Float16* __restrict__ w1h, const _Float16* __restrict__ w2h,
    const _Float16* __restrict__ w2t, const float* __restrict__ w1t,
    const float* __restrict__ conv1_b, const float* __restrict__ conv2_b,
    const float* __restrict__ deconv2_b, const float* __restrict__ deconv1_b,
    const float* __restrict__ lin_w, const float* __restrict__ lin_b,
    float* __restrict__ recon) {
  extern __shared__ char smem[];
  _Float16* s_in  = (_Float16*)(smem);               // [2][32][32]
  _Float16* act1t = (_Float16*)(smem + OFF_ACT1T);   // [625][32]  (pixel-major, ch-padded)
  _Float16* act2t = (_Float16*)(smem + OFF_ACT2T);   // [29*29][64] zero halo, ch-padded
  float*    cwt   = (float*)(smem + OFF_CWT);        // [64][24]  tap-major

  const int tid  = threadIdx.x;
  const int lane = tid & 31;
  const int wv   = tid >> 5;        // wave id 0..7
  const int hi   = lane >> 4;       // lane half
  const int col  = lane & 15;       // B/C column owned by this lane
  const int jb   = hi << 4;         // B fragment K sub-range base

  const int n   = blockIdx.x;
  const int gy0 = (n / NHG) * 16;
  const int gx0 = (n % NHG) * 16;

  __builtin_prefetch(w1h, 0, 0);
  __builtin_prefetch(w2h, 0, 0);
  __builtin_prefetch(w2t, 0, 0);

  // ---- stage 0a: zero act1t + act2t (padding must be finite: 0 * NaN = NaN) ----
  {
    unsigned* az1 = (unsigned*)act1t;   // 40000B = 10000 dwords
    for (int i = tid; i < 10000; i += 256) az1[i] = 0u;
    unsigned* az2 = (unsigned*)act2t;   // 107648B = 26912 dwords
    for (int i = tid; i < 26912; i += 256) az2[i] = 0u;
  }
  // ---- stage 0b: load 2x32x32 input patch -> LDS f16 ----
  for (int i = tid; i < 2048; i += 256) {
    int c = i >> 10, s = i & 1023, y = s >> 5, x = s & 31;
    const float* src = c ? x2 : x1;
    s_in[i] = (_Float16)src[(gy0 + y) * 1024 + (gx0 + x)];
  }
  __syncthreads();

  // ---- stage 1: conv1  M=24(32) N=625(640) K=128, ELU -> act1t ----
  for (int nt = wv; nt < 40; nt += 8) {
    const int s  = nt * 16 + col;
    const int sc = (s < 625) ? s : 624;
    const int oy = sc / 25, ox = sc % 25;
    const int sbase = oy * 32 + ox;
    v8f c0, c1;
#pragma unroll
    for (int v = 0; v < 8; ++v) {
      c0[v] = conv1_b[8 * hi + v];          // rows 0..15 always valid
      c1[v] = hi ? 0.f : conv1_b[16 + v];   // rows 16..23 valid, 24..31 padded
    }
#pragma unroll
    for (int kc = 0; kc < 4; ++kc) {
      v16h b;                               // gather (compiler vectorizes to b128s)
#pragma unroll
      for (int j = 0; j < 16; ++j) {
        int k = kc * 32 + jb + j;
        int kd = k >> 6, r = k & 63, ky = r >> 3, kx = r & 7;
        b[j] = s_in[kd * 1024 + sbase + ky * 32 + kx];
      }
      c0 = WMMA_F16(a_frag(w1h, 128, 0, kc * 32), b, c0);
      c1 = WMMA_F16(a_frag(w1h, 128, 16, kc * 32), b, c1);
    }
    if (s < 625) {
      _Float16* op = act1t + s * 32;
      v8h h0;
#pragma unroll
      for (int v = 0; v < 8; ++v) h0[v] = (_Float16)elu_f(c0[v]);
      *(v8h*)(op + 8 * hi) = h0;            // channels 8*hi .. 8*hi+7
      if (!hi) {
        v8h h1;
#pragma unroll
        for (int v = 0; v < 8; ++v) h1[v] = (_Float16)elu_f(c1[v]);
        *(v8h*)(op + 16) = h1;              // channels 16..23 (24..31 stay zero)
      }
    }
  }
  __syncthreads();

  // ---- stage 2: conv2  M=60(64) N=441(448) K=800 (25 taps x 32 ch), ReLU -> act2t ----
  // K-chunk outer / 4 output tiles inner: A-fragments loaded once per chunk.
  {
    v8f acc[4][4];                          // [mt][tile]
#pragma unroll
    for (int v = 0; v < 8; ++v) {
      int mb = 8 * hi + v;
      acc[0][0][v] = conv2_b[mb];
      acc[1][0][v] = conv2_b[mb + 16];
      acc[2][0][v] = conv2_b[mb + 32];
      acc[3][0][v] = (mb < 12) ? conv2_b[mb + 48] : 0.f;
    }
#pragma unroll
    for (int mt = 0; mt < 4; ++mt)
#pragma unroll
      for (int i = 1; i < 4; ++i) acc[mt][i] = acc[mt][0];

    int sb[4];                              // per-tile spatial base (clamped)
#pragma unroll
    for (int i = 0; i < 4; ++i) {
      int s  = (wv + 8 * i) * 16 + col;     // tiles 28..31 are phantom (never stored)
      int sx = (s < 441) ? s : 440;
      sb[i] = (sx / 21) * 25 + (sx % 21);
    }
    for (int kc = 0; kc < 25; ++kc) {       // one 5x5 tap per K-chunk (uniform offset)
      const int koff = (kc / 5) * 25 + (kc % 5);
      v16h b0 = b_frag_lds(act1t + ((sb[0] + koff) << 5) + jb);
      v16h b1 = b_frag_lds(act1t + ((sb[1] + koff) << 5) + jb);
      v16h b2 = b_frag_lds(act1t + ((sb[2] + koff) << 5) + jb);
      v16h b3 = b_frag_lds(act1t + ((sb[3] + koff) << 5) + jb);
#pragma unroll
      for (int mt = 0; mt < 4; ++mt) {
        v16h a = a_frag(w2h, 800, mt * 16, kc * 32);
        acc[mt][0] = WMMA_F16(a, b0, acc[mt][0]);
        acc[mt][1] = WMMA_F16(a, b1, acc[mt][1]);
        acc[mt][2] = WMMA_F16(a, b2, acc[mt][2]);
        acc[mt][3] = WMMA_F16(a, b3, acc[mt][3]);
      }
    }
#pragma unroll
    for (int i = 0; i < 4; ++i) {
      int s = (wv + 8 * i) * 16 + col;
      if (s < 441) {
        int oy = s / 21, ox = s % 21;
        _Float16* op = act2t + (((oy + 4) * 29 + (ox + 4)) << 6);
        v8h h;
#pragma unroll
        for (int v = 0; v < 8; ++v) h[v] = (_Float16)fmaxf(acc[0][i][v], 0.f);
        *(v8h*)(op + 8 * hi) = h;
#pragma unroll
        for (int v = 0; v < 8; ++v) h[v] = (_Float16)fmaxf(acc[1][i][v], 0.f);
        *(v8h*)(op + 16 + 8 * hi) = h;
#pragma unroll
        for (int v = 0; v < 8; ++v) h[v] = (_Float16)fmaxf(acc[2][i][v], 0.f);
        *(v8h*)(op + 32 + 8 * hi) = h;
        if (!hi) {
#pragma unroll
          for (int v = 0; v < 8; ++v) h[v] = (_Float16)fmaxf(acc[3][i][v], 0.f);
          *(v8h*)(op + 48) = h;             // channels 48..55
        } else {
#pragma unroll
          for (int v = 0; v < 4; ++v)       // channels 56..59 (60..63 stay zero)
            op[56 + v] = (_Float16)fmaxf(acc[3][i][v], 0.f);
        }
      }
    }
  }
  __syncthreads();

  // ---- stage 3: deconv2 as conv  M=24(32) N=625(640) K=1600 (25 taps x 64 ch), ELU -> act1t ----
  // K-chunk outer / 5 output tiles inner: A-fragments loaded once per chunk.
  {
    v8f c0[5], c1[5];
#pragma unroll
    for (int v = 0; v < 8; ++v) {
      c0[0][v] = deconv2_b[8 * hi + v];
      c1[0][v] = hi ? 0.f : deconv2_b[16 + v];
    }
#pragma unroll
    for (int i = 1; i < 5; ++i) { c0[i] = c0[0]; c1[i] = c1[0]; }

    int pb[5];
#pragma unroll
    for (int i = 0; i < 5; ++i) {
      int s  = (wv + 8 * i) * 16 + col;
      int sx = (s < 625) ? s : 624;
      pb[i] = (sx / 25) * 29 + (sx % 25);
    }
    for (int kc = 0; kc < 50; ++kc) {       // chunk = half a tap (32 of 64 channels)
      const int tap = kc >> 1;
      const int ocb = (kc & 1) << 5;
      const int toff = (tap / 5) * 29 + (tap % 5);
      v16h a0 = a_frag(w2t, 1600, 0, kc * 32);
      v16h a1 = a_frag(w2t, 1600, 16, kc * 32);
#pragma unroll
      for (int i = 0; i < 5; ++i) {
        v16h b = b_frag_lds(act2t + ((pb[i] + toff) << 6) + ocb + jb);
        c0[i] = WMMA_F16(a0, b, c0[i]);
        c1[i] = WMMA_F16(a1, b, c1[i]);
      }
    }
#pragma unroll
    for (int i = 0; i < 5; ++i) {
      int s = (wv + 8 * i) * 16 + col;
      if (s < 625) {
        _Float16* op = act1t + s * 32;
        v8h h0;
#pragma unroll
        for (int v = 0; v < 8; ++v) h0[v] = (_Float16)elu_f(c0[i][v]);
        *(v8h*)(op + 8 * hi) = h0;
        if (!hi) {
          v8h h1;
#pragma unroll
          for (int v = 0; v < 8; ++v) h1[v] = (_Float16)elu_f(c1[i][v]);
          *(v8h*)(op + 16) = h1;
        }
      }
    }
  }
  __syncthreads();

  // ---- stage 4: deconv1 with lin_w folded in, + lin_b, scatter-add to recon ----
  const float lw0 = lin_w[2 * n], lw1 = lin_w[2 * n + 1];
  for (int i = tid; i < 64 * 24; i += 256) {   // cwt[tap][ic]
    int kk = i / 24, ic = i % 24;
    cwt[i] = lw0 * w1t[ic * 128 + kk] + lw1 * w1t[ic * 128 + 64 + kk];
  }
  __syncthreads();
  const float base = lin_b[n] + (lw0 + lw1) * deconv1_b[0];
  for (int o = tid; o < 1024; o += 256) {
    int y = o >> 5, x = o & 31;
    float acc = base;
    int ky0 = (y < 7) ? 7 - y : 0;
    int ky1 = (y > 24) ? 31 - y : 7;
    int kx0 = (x < 7) ? 7 - x : 0;
    int kx1 = (x > 24) ? 31 - x : 7;
    for (int ky = ky0; ky <= ky1; ++ky) {
      int iy = y + ky - 7;
      for (int kx = kx0; kx <= kx1; ++kx) {
        int ix = x + kx - 7;
        const _Float16* ap = act1t + (iy * 25 + ix) * 32;   // 24 contiguous channels
        const float*    cp = cwt + (ky * 8 + kx) * 24;
#pragma unroll
        for (int ic = 0; ic < 24; ++ic) acc += (float)ap[ic] * cp[ic];
      }
    }
    atomicAdd(&recon[(gy0 + y) * 1024 + (gx0 + x)], acc);
  }
}

// ---------------- kernel 4: out = x2 - recon * linear1_w ----------------
__global__ void finalize_kernel(const float* __restrict__ x2, const float* __restrict__ recon,
                                const float* __restrict__ linear1_w, float* __restrict__ out) {
  int i = blockIdx.x * 256 + threadIdx.x;
  out[i] = x2[i] - recon[i] * linear1_w[0];
}

extern "C" void kernel_launch(void* const* d_in, const int* in_sizes, int n_in,
                              void* d_out, int out_size, void* d_ws, size_t ws_size,
                              hipStream_t stream) {
  (void)in_sizes; (void)n_in; (void)out_size; (void)ws_size;
  const float* x1        = (const float*)d_in[0];
  const float* x2        = (const float*)d_in[1];
  const float* conv1_w   = (const float*)d_in[2];
  const float* conv1_b   = (const float*)d_in[3];
  const float* conv2_w   = (const float*)d_in[4];
  const float* conv2_b   = (const float*)d_in[5];
  const float* deconv2_w = (const float*)d_in[6];
  const float* deconv2_b = (const float*)d_in[7];
  const float* deconv1_w = (const float*)d_in[8];
  const float* deconv1_b = (const float*)d_in[9];
  const float* lin_w     = (const float*)d_in[10];
  const float* lin_b     = (const float*)d_in[11];
  const float* linear1_w = (const float*)d_in[12];

  char* ws = (char*)d_ws;
  float*    recon = (float*)(ws + WS_RECON);
  _Float16* w1h   = (_Float16*)(ws + WS_W1H);
  _Float16* w2h   = (_Float16*)(ws + WS_W2H);
  _Float16* w2t   = (_Float16*)(ws + WS_W2T);
  float*    w1t   = (float*)(ws + WS_W1T);

  zero_recon<<<4096, 256, 0, stream>>>(recon);
  prep_weights<<<120, 256, 0, stream>>>(conv1_w, conv2_w, deconv2_w, deconv1_w,
                                        w1h, w2h, w2t, w1t);
  patch_pipeline<<<NPATCH, 256, SMEM_BYTES, stream>>>(
      x1, x2, w1h, w2h, w2t, w1t,
      conv1_b, conv2_b, deconv2_b, deconv1_b, lin_w, lin_b, recon);
  finalize_kernel<<<4096, 256, 0, stream>>>(x2, recon, linear1_w, (float*)d_out);
}
